// W4A8AWQRefGatedMLPFusedMoE_47562467836587
// MI455X (gfx1250) — 8-bit matrix operand order repaired and checked by reading, compiled, not run on hardware
//
#include <hip/hip_runtime.h>
#include <hip/hip_bf16.h>
#include <math.h>

#define E_ 8
#define H_ 2048
#define I_ 5632
#define G_ 128
#define T_ 256

typedef int   v16i __attribute__((ext_vector_type(16)));
typedef float v8f  __attribute__((ext_vector_type(8)));

#define LDSPITCH 144
#define TILEBYTES (16 * LDSPITCH)

#if __has_builtin(__builtin_amdgcn_cvt_pk_fp8_f32)
#define HAVE_HW_FP8 1
#endif

__device__ __forceinline__ unsigned int int8_to_e4m3_lut(int w) {
  const unsigned long long lo = 0x4E4C4A4844403800ull;
  const unsigned long long hi = 0xB8C0C4C8CACCCED0ull;
  int idx               = w & 15;
  unsigned long long tb = (idx & 8) ? hi : lo;
  return (unsigned int)((tb >> ((idx & 7) * 8)) & 0xFFu);
}

__device__ __forceinline__ unsigned int pack4_e4m3(unsigned int b0,
                                                   unsigned int b1,
                                                   unsigned int b2,
                                                   unsigned int b3) {
#ifdef HAVE_HW_FP8
  float f0 = (float)(int)(signed char)b0;
  float f1 = (float)(int)(signed char)b1;
  float f2 = (float)(int)(signed char)b2;
  float f3 = (float)(int)(signed char)b3;
  int d    = __builtin_amdgcn_cvt_pk_fp8_f32(f0, f1, 0, false);
  d        = __builtin_amdgcn_cvt_pk_fp8_f32(f2, f3, d, true);
  return (unsigned int)d;
#else
  return int8_to_e4m3_lut((signed char)b0) |
         (int8_to_e4m3_lut((signed char)b1) << 8) |
         (int8_to_e4m3_lut((signed char)b2) << 16) |
         (int8_to_e4m3_lut((signed char)b3) << 24);
#endif
}

__device__ __forceinline__ void stage_rows4(const uint4& r0, const uint4& r1,
                                            const uint4& r2, const uint4& r3,
                                            unsigned char* dstBase,
                                            int srow4) {
  const unsigned char* p0 = (const unsigned char*)&r0;
  const unsigned char* p1 = (const unsigned char*)&r1;
  const unsigned char* p2 = (const unsigned char*)&r2;
  const unsigned char* p3 = (const unsigned char*)&r3;
#pragma unroll
  for (int n = 0; n < 16; ++n) {
    unsigned int d = pack4_e4m3(p0[n], p1[n], p2[n], p3[n]);
    *(unsigned int*)(dstBase + n * LDSPITCH + srow4 * 4) = d;
  }
}

__device__ __forceinline__ unsigned char f32_to_e4m3_sw(float f) {
  unsigned int x    = __float_as_uint(f);
  unsigned int sign = (x >> 31) << 7;
  unsigned int absx = x & 0x7FFFFFFFu;
  if (absx > 0x7F800000u) return (unsigned char)(sign | 0x7F);
  if (__uint_as_float(absx) > 448.f) absx = 0x43E00000u;
  int e          = (int)(absx >> 23) - 127;
  unsigned int m = absx & 0x7FFFFFu;
  unsigned int out;
  if (e < -6) {
    int shift = -6 - e;
    if (shift > 4) return (unsigned char)sign;
    unsigned int full = m | 0x800000u;
    int rs            = 20 + shift;
    unsigned int kept = full >> rs;
    unsigned int rem  = full & ((1u << rs) - 1u);
    unsigned int half = 1u << (rs - 1);
    if (rem > half || (rem == half && (kept & 1u))) kept++;
    out = kept;
  } else {
    unsigned int kept = m >> 20;
    unsigned int rem  = m & 0xFFFFFu;
    if (rem > 0x80000u || (rem == 0x80000u && (kept & 1u))) kept++;
    int ee = e + 7;
    if (kept == 8u) { kept = 0u; ee++; }
    if (ee > 15 || (ee == 15 && kept > 6u)) { ee = 15; kept = 6u; }
    out = ((unsigned)ee << 3) | kept;
  }
  return (unsigned char)(sign | out);
}

__device__ __forceinline__ unsigned char f32_to_e4m3(float v) {
  v = fminf(fmaxf(v, -448.f), 448.f);
#ifdef HAVE_HW_FP8
  return (unsigned char)(__builtin_amdgcn_cvt_pk_fp8_f32(v, v, 0, false) & 0xFF);
#else
  return f32_to_e4m3_sw(v);
#endif
}

__global__ void moe_routing_kernel(const float* __restrict__ logits,
                                   float* __restrict__ gate) {
  int t = blockIdx.x * blockDim.x + threadIdx.x;
  if (t >= T_) return;
  float lg[E_];
#pragma unroll
  for (int e = 0; e < E_; ++e) lg[e] = logits[t * E_ + e];
  int i0 = 0, i1 = -1;
  float v0 = lg[0], v1 = -INFINITY;
#pragma unroll
  for (int e = 1; e < E_; ++e) {
    float v = lg[e];
    if (v > v0)      { v1 = v0; i1 = i0; v0 = v; i0 = e; }
    else if (v > v1) { v1 = v;  i1 = e; }
  }
  float g0 = 1.f / (1.f + __expf(v1 - v0));
  float g1 = 1.f - g0;
#pragma unroll
  for (int e = 0; e < E_; ++e)
    gate[t * E_ + e] = (e == i0) ? g0 : (e == i1 ? g1 : 0.f);
}

__global__ void moe_quant_act_kernel(const float* __restrict__ hs,
                                     const float* __restrict__ s1,
                                     unsigned char* __restrict__ actq) {
  const long long npairs = (long long)E_ * T_ * H_ / 2;
  for (long long p = blockIdx.x * (long long)blockDim.x + threadIdx.x;
       p < npairs; p += (long long)gridDim.x * blockDim.x) {
    long long i  = p * 2;
    int e        = (int)(i / ((long long)T_ * H_));
    long long th = i % ((long long)T_ * H_);
    float s      = s1[e];
    float a      = fminf(fmaxf(hs[th] / s, -448.f), 448.f);
    float b      = fminf(fmaxf(hs[th + 1] / s, -448.f), 448.f);
#ifdef HAVE_HW_FP8
    int pk = __builtin_amdgcn_cvt_pk_fp8_f32(a, b, 0, false);
    *(unsigned short*)(actq + i) = (unsigned short)(pk & 0xFFFF);
#else
    actq[i]     = f32_to_e4m3_sw(a);
    actq[i + 1] = f32_to_e4m3_sw(b);
#endif
  }
}

__global__ __launch_bounds__(512) void moe_fc1_kernel(
    const unsigned char* __restrict__ actq,
    const signed char* __restrict__ w31,
    const float* __restrict__ w31s,
    const float* __restrict__ s1v,
    const float* __restrict__ s2v,
    unsigned char* __restrict__ hq)
{
  __shared__ __align__(16) unsigned char sB[2 * 2 * TILEBYTES];

  const int nblk = I_ / 16;
  const int e    = blockIdx.x / nblk;
  const int n0   = (blockIdx.x % nblk) * 16;
  const int tid  = threadIdx.x;
  const int lane = tid & 31;
  const int wave = tid >> 5;
  const int half = lane >> 4;
  const int col  = lane & 15;
  const int m0   = wave * 16;
  const int KSTEPS = H_ / G_;

  const float s1 = s1v[e];
  const float s2 = s2v[e];

  v8f acc_u    = {0.f, 0.f, 0.f, 0.f, 0.f, 0.f, 0.f, 0.f};
  v8f acc_g    = {0.f, 0.f, 0.f, 0.f, 0.f, 0.f, 0.f, 0.f};
  const v8f zc = {0.f, 0.f, 0.f, 0.f, 0.f, 0.f, 0.f, 0.f};

  const int stile = tid >> 5;
  const int srow4 = tid & 31;
  const size_t ld2I = 2 * (size_t)I_;
  const signed char* wrow =
      w31 + ((size_t)e * H_ + srow4 * 4) * ld2I + n0 + (size_t)stile * I_;
  unsigned char* sdst0 = sB + stile * TILEBYTES;

  const unsigned char* aRow =
      actq + ((size_t)e * T_ + (m0 + col)) * H_ + half * 8;

  uint4 r0, r1, r2, r3;
  if (tid < 64) {
    r0 = *(const uint4*)(wrow);
    r1 = *(const uint4*)(wrow + ld2I);
    r2 = *(const uint4*)(wrow + 2 * ld2I);
    r3 = *(const uint4*)(wrow + 3 * ld2I);
    stage_rows4(r0, r1, r2, r3, sdst0, srow4);
  }

#pragma unroll 1
  for (int kg = 0; kg < KSTEPS; ++kg) {
    if (tid < 64 && kg + 1 < KSTEPS) {
      const signed char* src = wrow + (size_t)(kg + 1) * 128 * ld2I;
      r0 = *(const uint4*)(src);
      r1 = *(const uint4*)(src + ld2I);
      r2 = *(const uint4*)(src + 2 * ld2I);
      r3 = *(const uint4*)(src + 3 * ld2I);
      if (kg + 2 < KSTEPS) __builtin_prefetch(src + 128 * ld2I, 0, 3);
    }
    __syncthreads();

    const unsigned char* buf = sB + (kg & 1) * 2 * TILEBYTES;
    const int k0             = kg * 128;

    v16i A;
#pragma unroll
    for (int j = 0; j < 8; ++j) {
      uint2 t      = *(const uint2*)(aRow + k0 + 32 * (j >> 1) + 8 * (j & 1) + half * 8);
      A[2 * j]     = (int)t.x;
      A[2 * j + 1] = (int)t.y;
    }
    v16i Bu, Bg;
#pragma unroll
    for (int g = 0; g < 4; ++g) {
      uint4 tu = *(const uint4*)(buf + col * LDSPITCH + g * 32 + half * 16);
      uint4 tg = *(const uint4*)(buf + TILEBYTES + col * LDSPITCH + g * 32 +
                                 half * 16);
      Bu[4 * g + 0] = (int)tu.x; Bu[4 * g + 1] = (int)tu.y;
      Bu[4 * g + 2] = (int)tu.z; Bu[4 * g + 3] = (int)tu.w;
      Bg[4 * g + 0] = (int)tg.x; Bg[4 * g + 1] = (int)tg.y;
      Bg[4 * g + 2] = (int)tg.z; Bg[4 * g + 3] = (int)tg.w;
    }

    v8f pu = __builtin_amdgcn_wmma_f32_16x16x128_fp8_fp8(A, Bu, (short)0, zc,
                                                         false, false);
    v8f pg = __builtin_amdgcn_wmma_f32_16x16x128_fp8_fp8(A, Bg, (short)0, zc,
                                                         false, false);

    const float su =
        w31s[((size_t)e * (H_ / G_) + kg) * ld2I + n0 + col];
    const float sg =
        w31s[((size_t)e * (H_ / G_) + kg) * ld2I + I_ + n0 + col];
#pragma unroll
    for (int i = 0; i < 8; ++i) {
      acc_u[i] += pu[i] * su;
      acc_g[i] += pg[i] * sg;
    }

    if (tid < 64 && kg + 1 < KSTEPS)
      stage_rows4(r0, r1, r2, r3, sdst0 + ((kg + 1) & 1) * 2 * TILEBYTES,
                  srow4);
  }

#pragma unroll
  for (int i = 0; i < 8; ++i) {
    float up  = acc_u[i] * s1;
    float gt  = acc_g[i] * s1;
    float sig = 1.f / (1.f + __expf(-gt));
    float h   = up * (gt * sig);
    int m     = m0 + i + 8 * half;
    hq[((size_t)e * T_ + m) * I_ + n0 + col] = f32_to_e4m3(h / s2);
  }
}

__global__ __launch_bounds__(512) void moe_fc2_kernel(
    const unsigned char* __restrict__ hq,
    const signed char* __restrict__ w2,
    const float* __restrict__ w2s,
    const float* __restrict__ s2v,
    const float* __restrict__ gate,
    float* __restrict__ out)
{
  __shared__ __align__(16) unsigned char sB[2 * TILEBYTES];

  const int n0   = blockIdx.x * 16;
  const int tid  = threadIdx.x;
  const int lane = tid & 31;
  const int wave = tid >> 5;
  const int half = lane >> 4;
  const int col  = lane & 15;
  const int m0   = wave * 16;
  const int KSTEPS = I_ / G_;

  const v8f zc = {0.f, 0.f, 0.f, 0.f, 0.f, 0.f, 0.f, 0.f};
  float acc[8] = {0.f, 0.f, 0.f, 0.f, 0.f, 0.f, 0.f, 0.f};

  const int srow4 = tid & 31;

#pragma unroll 1
  for (int e = 0; e < E_; ++e) {
    v8f eacc = {0.f, 0.f, 0.f, 0.f, 0.f, 0.f, 0.f, 0.f};
    const unsigned char* aRow =
        hq + ((size_t)e * T_ + (m0 + col)) * I_ + half * 8;
    const signed char* wrow =
        w2 + ((size_t)e * I_ + srow4 * 4) * (size_t)H_ + n0;

    uint4 r0, r1, r2, r3;
    if (tid < 32) {
      r0 = *(const uint4*)(wrow);
      r1 = *(const uint4*)(wrow + H_);
      r2 = *(const uint4*)(wrow + 2 * H_);
      r3 = *(const uint4*)(wrow + 3 * H_);
      stage_rows4(r0, r1, r2, r3, sB, srow4);
    }

#pragma unroll 1
    for (int kg = 0; kg < KSTEPS; ++kg) {
      if (tid < 32 && kg + 1 < KSTEPS) {
        const signed char* src = wrow + (size_t)(kg + 1) * 128 * (size_t)H_;
        r0 = *(const uint4*)(src);
        r1 = *(const uint4*)(src + H_);
        r2 = *(const uint4*)(src + 2 * H_);
        r3 = *(const uint4*)(src + 3 * H_);
        if (kg + 2 < KSTEPS)
          __builtin_prefetch(src + (size_t)128 * H_, 0, 3);
      }
      __syncthreads();

      const unsigned char* buf = sB + (kg & 1) * TILEBYTES;
      const int k0             = kg * 128;

      v16i A, B;
#pragma unroll
      for (int j = 0; j < 8; ++j) {
        uint2 t      = *(const uint2*)(aRow + k0 + 32 * (j >> 1) + 8 * (j & 1) + half * 8);
        A[2 * j]     = (int)t.x;
        A[2 * j + 1] = (int)t.y;
      }
#pragma unroll
      for (int g = 0; g < 4; ++g) {
        uint4 tb = *(const uint4*)(buf + col * LDSPITCH + g * 32 + half * 16);
        B[4 * g + 0] = (int)tb.x; B[4 * g + 1] = (int)tb.y;
        B[4 * g + 2] = (int)tb.z; B[4 * g + 3] = (int)tb.w;
      }

      v8f p = __builtin_amdgcn_wmma_f32_16x16x128_fp8_fp8(A, B, (short)0, zc,
                                                          false, false);
      const float ws =
          w2s[((size_t)e * (I_ / G_) + kg) * (size_t)H_ + n0 + col];
#pragma unroll
      for (int i = 0; i < 8; ++i) eacc[i] += p[i] * ws;

      if (tid < 32 && kg + 1 < KSTEPS)
        stage_rows4(r0, r1, r2, r3, sB + ((kg + 1) & 1) * TILEBYTES, srow4);
    }

    const float s2 = s2v[e];
#pragma unroll
    for (int i = 0; i < 8; ++i) {
      int m = m0 + i + 8 * half;
      acc[i] += eacc[i] * s2 * gate[m * E_ + e];
    }
  }

#pragma unroll
  for (int i = 0; i < 8; ++i) {
    int m = m0 + i + 8 * half;
    out[(size_t)m * H_ + n0 + col] = acc[i];
  }
}

extern "C" void kernel_launch(void* const* d_in, const int* in_sizes, int n_in,
                              void* d_out, int out_size, void* d_ws,
                              size_t ws_size, hipStream_t stream) {
  const float*       hs     = (const float*)d_in[0];
  const float*       logits = (const float*)d_in[1];
  const signed char* w31    = (const signed char*)d_in[2];
  const signed char* w2     = (const signed char*)d_in[3];
  const float*       w31s   = (const float*)d_in[4];
  const float*       w2s    = (const float*)d_in[5];
  const float*       s1     = (const float*)d_in[6];
  const float*       s2     = (const float*)d_in[7];
  float*             out    = (float*)d_out;

  unsigned char* ws   = (unsigned char*)d_ws;
  float*         gate = (float*)ws;
  unsigned char* actq = ws + 8192;
  unsigned char* hq   = actq + (size_t)E_ * T_ * H_;

  moe_routing_kernel<<<1, 256, 0, stream>>>(logits, gate);
  moe_quant_act_kernel<<<2048, 256, 0, stream>>>(hs, s1, actq);
  moe_fc1_kernel<<<E_ * (I_ / 16), 512, 0, stream>>>(actq, w31, w31s, s1, s2,
                                                     hq);
  moe_fc2_kernel<<<H_ / 16, 512, 0, stream>>>(hq, w2, w2s, s2, gate, out);
}
